// DecoderLayer_70205535421363
// MI455X (gfx1250) — matrix operand order repaired and checked by reading, not compiled, not run on hardware
//
#include <hip/hip_runtime.h>

// ---------------------------------------------------------------------------
// Types for WMMA
// ---------------------------------------------------------------------------
typedef __attribute__((ext_vector_type(16))) __bf16        v16bf;
typedef __attribute__((ext_vector_type(8)))  float         v8f;
typedef __attribute__((ext_vector_type(4)))  unsigned int  u32x4;

union Frag16 { u32x4 u[2]; v16bf v; };

__device__ __forceinline__ unsigned short f32_to_bf16(float f) {
  unsigned int u = __float_as_uint(f);
  u += 0x7FFFu + ((u >> 16) & 1u);            // round-to-nearest-even
  return (unsigned short)(u >> 16);
}

__device__ __forceinline__ float fast_rcp(float x) {
#if __has_builtin(__builtin_amdgcn_rcpf)
  return __builtin_amdgcn_rcpf(x);
#else
  return 1.0f / x;
#endif
}

// ---------------------------------------------------------------------------
// CDNA5 async global->LDS copies (ASYNCcnt-tracked). Async loads complete
// in order, so a partial wait (s_wait_asynccnt N) retires the oldest copies
// — this is what enables the 2-deep LDS pipeline below.
// ---------------------------------------------------------------------------
__device__ __forceinline__ void async_load_b128(const void* g, void* l) {
  unsigned lds_addr = (unsigned)(unsigned long long)(size_t)l;
  unsigned long long gaddr = (unsigned long long)(size_t)g;
  asm volatile("global_load_async_to_lds_b128 %0, %1, off"
               :: "v"(lds_addr), "v"(gaddr) : "memory");
}
__device__ __forceinline__ void wait_async0()  { asm volatile("s_wait_asynccnt 0x0" ::: "memory"); }
__device__ __forceinline__ void wait_async4()  { asm volatile("s_wait_asynccnt 0x4" ::: "memory"); }
__device__ __forceinline__ void wait_async8()  { asm volatile("s_wait_asynccnt 0x8" ::: "memory"); }

// ---------------------------------------------------------------------------
// fp32 -> bf16 cast (weights)
// ---------------------------------------------------------------------------
__global__ __launch_bounds__(256) void cast_f32_bf16_kernel(
    const float* __restrict__ in, unsigned short* __restrict__ out, size_t n) {
  size_t i = (size_t)blockIdx.x * blockDim.x + threadIdx.x;
  size_t stride = (size_t)gridDim.x * blockDim.x;
  for (; i < n; i += stride) out[i] = f32_to_bf16(in[i]);
}

// ---------------------------------------------------------------------------
// RMSNorm over rows of length `rowlen` (f32 in, bf16 out). One block per row.
// ---------------------------------------------------------------------------
__global__ __launch_bounds__(256) void rmsnorm_row_kernel(
    const float* __restrict__ x, const float* __restrict__ scale,
    unsigned short* __restrict__ out, int rowlen) {
  __shared__ float red[256];
  const int row = blockIdx.x;
  const float* xr = x + (size_t)row * rowlen;
  float ss = 0.0f;
  for (int i = threadIdx.x; i < rowlen; i += 256) { float v = xr[i]; ss += v * v; }
  red[threadIdx.x] = ss;
  __syncthreads();
  for (int s = 128; s > 0; s >>= 1) {
    if (threadIdx.x < s) red[threadIdx.x] += red[threadIdx.x + s];
    __syncthreads();
  }
  float inv = rsqrtf(red[0] / (float)rowlen + 1e-6f);
  for (int i = threadIdx.x; i < rowlen; i += 256)
    out[(size_t)row * rowlen + i] = f32_to_bf16(xr[i] * inv * scale[i]);
}

// Per-head RMSNorm: rows of 128. One block (128 threads) per row.
__global__ __launch_bounds__(128) void headnorm_kernel(
    const float* __restrict__ x, const float* __restrict__ scale,
    unsigned short* __restrict__ out) {
  __shared__ float red[128];
  size_t row = blockIdx.x;
  float v = x[row * 128 + threadIdx.x];
  red[threadIdx.x] = v * v;
  __syncthreads();
  for (int s = 64; s > 0; s >>= 1) {
    if (threadIdx.x < s) red[threadIdx.x] += red[threadIdx.x + s];
    __syncthreads();
  }
  float inv = rsqrtf(red[0] / 128.0f + 1e-6f);
  out[row * 128 + threadIdx.x] = f32_to_bf16(v * inv * scale[threadIdx.x]);
}

// ---------------------------------------------------------------------------
// bf16 WMMA GEMM: C[M,N] = A[M,K] x B[K,N], A/B row-major bf16, acc f32.
// Block tile 128x128x64, 8 waves, wave tile 32x64 (2x4 WMMA), 2 K-steps.
// 2-deep async pipeline: tile i+1's global->LDS copies run during tile i's
// WMMAs; s_wait_asynccnt 8 retires only the current tile's copies.
// Epilogue modes:
//   0: out f32 = acc
//   1: out bf16 = acc
//   2: out bf16 = silu(aux) * acc       (gated MLP)
//   3: out f32 = acc + aux              (residual add)
// ---------------------------------------------------------------------------
#define TBM 128
#define TBN 128
#define TBK 64
#define LDSA 72   // [row][k] stride (bf16 elems), 144B rows -> 16B aligned
#define LDSB 72   // [col][k] stride
#define BST_STR 136

__global__ __launch_bounds__(256) void gemm_bf16_wmma(
    const unsigned short* __restrict__ A,
    const unsigned short* __restrict__ B,
    const float* __restrict__ aux,
    void* __restrict__ outp,
    int M, int N, int K, int mode) {
  __shared__ __align__(16) unsigned short As[2][TBM * LDSA];        // [row][k], ping-pong
  __shared__ __align__(16) unsigned short Bs[TBN * LDSB];           // [col][k] (transposed)
  __shared__ __align__(16) unsigned short Bstage[2][TBK * BST_STR]; // [k][n] staging, ping-pong

  const int tid      = threadIdx.x;
  const int lane     = tid & 31;
  const int wave     = tid >> 5;
  const int lane16   = lane & 15;
  const int laneHalf = lane >> 4;
  const int waveM    = (wave >> 1) * 32;
  const int waveN    = (wave & 1) * 64;
  const int mBase    = blockIdx.y * TBM;
  const int nBase    = blockIdx.x * TBN;

  v8f acc[2][4];
#pragma unroll
  for (int i = 0; i < 2; ++i)
#pragma unroll
    for (int j = 0; j < 4; ++j)
#pragma unroll
      for (int e = 0; e < 8; ++e) acc[i][j][e] = 0.0f;

  // Issue 8 async b128 copies (4 A-chunks + 4 B-chunks) for one K-tile.
  auto issue_tile = [&](int k0, int buf) {
#pragma unroll
    for (int c = 0; c < 4; ++c) {
      int chunk = tid * 4 + c;              // 0..1023
      int row = chunk >> 3;                 // 0..127
      int col = (chunk & 7) << 3;           // 0..56
      async_load_b128(A + (size_t)(mBase + row) * K + k0 + col,
                      &As[buf][row * LDSA + col]);
    }
#pragma unroll
    for (int c = 0; c < 4; ++c) {
      int chunk = tid * 4 + c;
      int krow = chunk >> 4;                // 0..63
      int ncol = (chunk & 15) << 3;         // 0..120
      async_load_b128(B + (size_t)(k0 + krow) * N + nBase + ncol,
                      &Bstage[buf][krow * BST_STR + ncol]);
    }
  };

  issue_tile(0, 0);
  int p = 0;

  for (int k0 = 0; k0 < K; k0 += TBK) {
    const bool more = (k0 + TBK) < K;
    if (more) {
      if (k0 + 2 * TBK < K) {  // L2 prefetch two tiles ahead (global_prefetch_b8)
        __builtin_prefetch(A + (size_t)(mBase + (tid >> 1)) * K + k0 + 2 * TBK, 0, 1);
        __builtin_prefetch(B + (size_t)(k0 + 2 * TBK + (tid >> 3)) * N + nBase, 0, 1);
      }
      issue_tile(k0 + TBK, p ^ 1);   // overlap next tile's copies with this tile's math
      wait_async8();                 // retire current tile's 8 copies only
    } else {
      wait_async0();
    }

    // Transpose the chunks THIS thread staged (wave-local data): [k][n]->[n][k]
#pragma unroll
    for (int c = 0; c < 4; ++c) {
      int chunk = tid * 4 + c;
      int krow = chunk >> 4;
      int ncol = (chunk & 15) << 3;
      u32x4 val = *(const u32x4*)(&Bstage[p][krow * BST_STR + ncol]);
      const unsigned short* vv = (const unsigned short*)&val;
#pragma unroll
      for (int j = 0; j < 8; ++j) Bs[(ncol + j) * LDSB + krow] = vv[j];
    }
    __syncthreads();  // all waves' As[p] / Bs visible

#pragma unroll
    for (int dk = 0; dk < 2; ++dk) {
      Frag16 af[2], bfr[4];
#pragma unroll
      for (int mt = 0; mt < 2; ++mt) {
        const unsigned short* fp =
            &As[p][(waveM + mt * 16 + lane16) * LDSA + dk * 32 + laneHalf * 8];
        af[mt].u[0] = *(const u32x4*)(fp);
        af[mt].u[1] = *(const u32x4*)(fp + 16);
      }
#pragma unroll
      for (int nt = 0; nt < 4; ++nt) {
        const unsigned short* fp =
            Bs + (waveN + nt * 16 + lane16) * LDSB + dk * 32 + laneHalf * 8;
        bfr[nt].u[0] = *(const u32x4*)(fp);
        bfr[nt].u[1] = *(const u32x4*)(fp + 16);
      }
#pragma unroll
      for (int mt = 0; mt < 2; ++mt)
#pragma unroll
        for (int nt = 0; nt < 4; ++nt)
          acc[mt][nt] = __builtin_amdgcn_wmma_f32_16x16x32_bf16(
              false, af[mt].v, false, bfr[nt].v, (short)0, acc[mt][nt], false, false);
    }
    __syncthreads();  // LDS reads done before next iteration's writes
    p ^= 1;
  }

  // Epilogue. C layout: VGPR j -> row j (lanes 0-15) / j+8 (lanes 16-31), col=lane16.
#pragma unroll
  for (int mt = 0; mt < 2; ++mt) {
#pragma unroll
    for (int nt = 0; nt < 4; ++nt) {
#pragma unroll
      for (int j = 0; j < 8; ++j) {
        int row = mBase + waveM + mt * 16 + j + laneHalf * 8;
        int col = nBase + waveN + nt * 16 + lane16;
        size_t idx = (size_t)row * N + col;
        float v = acc[mt][nt][j];
        if (mode == 0) {
          ((float*)outp)[idx] = v;
        } else if (mode == 1) {
          ((unsigned short*)outp)[idx] = f32_to_bf16(v);
        } else if (mode == 2) {
          float g = aux[idx];
          float s = g * fast_rcp(1.0f + __expf(-g));   // silu via v_rcp_f32
          ((unsigned short*)outp)[idx] = f32_to_bf16(s * v);
        } else {
          ((float*)outp)[idx] = v + aux[idx];
        }
      }
    }
  }
}

// ---------------------------------------------------------------------------
// Flash attention (causal, no scale — T5 style), WMMA QK^T and P*V.
// Q tile 64 rows, K tile 32 keys, HD = 128. One block = (b, h, q-tile).
// q/k/v/out layout: [B, S, H, HD] bf16. K/V tiles in a 2-deep async pipeline.
// ---------------------------------------------------------------------------
#define QT 64
#define KT 32
#define QS_STR 136
#define KS_STR 136
#define VST_STR 136
#define VS_STR 40
#define SS_STR 33
#define PS_STR 40

__global__ __launch_bounds__(256) void flash_attn_wmma(
    const unsigned short* __restrict__ Qg,
    const unsigned short* __restrict__ Kg,
    const unsigned short* __restrict__ Vg,
    unsigned short* __restrict__ Og,
    int Bn, int Sn, int Hn, int HDn) {
  __shared__ __align__(16) unsigned short Qs[QT * QS_STR];          // [q][d]
  __shared__ __align__(16) unsigned short Ks[2][KT * KS_STR];       // [key][d], ping-pong
  __shared__ __align__(16) unsigned short Vstage[2][KT * VST_STR];  // [key][d], ping-pong
  __shared__ __align__(16) unsigned short Vts[128 * VS_STR];        // [d][key]
  __shared__ float Ss[QT * SS_STR];                                 // scores f32
  __shared__ __align__(16) unsigned short Ps[QT * PS_STR];          // probs bf16
  __shared__ float m_s[QT], l_s[QT], alpha_s[QT];

  const int tid      = threadIdx.x;
  const int lane     = tid & 31;
  const int wave     = tid >> 5;
  const int lane16   = lane & 15;
  const int laneHalf = lane >> 4;

  const int bh    = blockIdx.x;
  const int bb    = bh / Hn;
  const int hh    = bh % Hn;
  const int qbase = blockIdx.y * QT;

  const size_t rowStride = (size_t)Hn * HDn;
  const size_t headOff   = (size_t)hh * HDn;
  const unsigned short* Qp = Qg + (size_t)bb * Sn * rowStride + headOff;
  const unsigned short* Kp = Kg + (size_t)bb * Sn * rowStride + headOff;
  const unsigned short* Vp = Vg + (size_t)bb * Sn * rowStride + headOff;

  // Async-load Q tile (64 x 128) — issued first, completes first (in-order).
#pragma unroll
  for (int c = 0; c < 4; ++c) {
    int chunk = tid * 4 + c;              // 0..1023
    int row = chunk >> 4;
    int dc = (chunk & 15) << 3;
    async_load_b128(Qp + (size_t)(qbase + row) * rowStride + dc,
                    Qs + row * QS_STR + dc);
  }
  if (tid < QT) { m_s[tid] = -1e30f; l_s[tid] = 0.0f; }

  // Issue 4 async copies (2 K-chunks + 2 V-chunks) for one key tile.
  auto issue_kv = [&](int kb, int buf) {
#pragma unroll
    for (int c = 0; c < 2; ++c) {
      int chunk = tid * 2 + c;
      int row = chunk >> 4;
      int dc = (chunk & 15) << 3;
      async_load_b128(Kp + (size_t)(kb + row) * rowStride + dc,
                      &Ks[buf][row * KS_STR + dc]);
    }
#pragma unroll
    for (int c = 0; c < 2; ++c) {
      int chunk = tid * 2 + c;
      int key = chunk >> 4;
      int dc = (chunk & 15) << 3;
      async_load_b128(Vp + (size_t)(kb + key) * rowStride + dc,
                      &Vstage[buf][key * VST_STR + dc]);
    }
  };

  const int waveMs = (wave >> 1) * 16;  // S-phase: 4x2 grid of 16x16 tiles
  const int waveNs = (wave & 1) * 16;
  const int waveMo = (wave >> 1) * 16;  // O-phase: 16 rows x 64 cols per wave
  const int waveNo = (wave & 1) * 64;

  v8f oacc[4];
#pragma unroll
  for (int nt = 0; nt < 4; ++nt)
#pragma unroll
    for (int e = 0; e < 8; ++e) oacc[nt][e] = 0.0f;

  const int nkt = (qbase + QT) / KT;   // causal: keys up to last query of tile
  issue_kv(0, 0);
  int p = 0;

  for (int kt = 0; kt < nkt; ++kt) {
    const int kb = kt * KT;
    if (kt + 1 < nkt) {
      issue_kv(kb + KT, p ^ 1);  // overlap next K/V copies with this tile's math
      wait_async4();             // retire Q (first iter) + current K/V copies
    } else {
      wait_async0();
    }

    // Transpose this thread's own staged V chunks -> [d][key]
#pragma unroll
    for (int c = 0; c < 2; ++c) {
      int chunk = tid * 2 + c;
      int key = chunk >> 4;
      int dc = (chunk & 15) << 3;
      u32x4 val = *(const u32x4*)(&Vstage[p][key * VST_STR + dc]);
      const unsigned short* vv = (const unsigned short*)&val;
#pragma unroll
      for (int j = 0; j < 8; ++j) Vts[(dc + j) * VS_STR + key] = vv[j];
    }
    __syncthreads();  // (A) Q/Ks[p]/Vts tiles visible

    // ---- S = Q * K^T : one 16x16 tile per wave, contraction over HD=128 ----
    v8f sacc;
#pragma unroll
    for (int e = 0; e < 8; ++e) sacc[e] = 0.0f;
#pragma unroll
    for (int dk = 0; dk < 4; ++dk) {
      Frag16 qa, kf;
      const unsigned short* qp = Qs + (waveMs + lane16) * QS_STR + dk * 32 + laneHalf * 8;
      qa.u[0] = *(const u32x4*)(qp);
      qa.u[1] = *(const u32x4*)(qp + 16);
      const unsigned short* kp = &Ks[p][(waveNs + lane16) * KS_STR + dk * 32 + laneHalf * 8];
      kf.u[0] = *(const u32x4*)(kp);
      kf.u[1] = *(const u32x4*)(kp + 16);
      sacc = __builtin_amdgcn_wmma_f32_16x16x32_bf16(
          false, qa.v, false, kf.v, (short)0, sacc, false, false);
    }
#pragma unroll
    for (int j = 0; j < 8; ++j)
      Ss[(waveMs + j + laneHalf * 8) * SS_STR + waveNs + lane16] = sacc[j];
    __syncthreads();  // (B) scores visible

    // ---- online softmax row pass (threads 0..63, one row each) ----
    if (tid < QT) {
      int qglob = qbase + tid;
      float mold = m_s[tid];
      float mx = mold;
      for (int j = 0; j < KT; ++j)
        if (kb + j <= qglob) mx = fmaxf(mx, Ss[tid * SS_STR + j]);
      float alpha = __expf(mold - mx);
      float lsum = 0.0f;
      for (int j = 0; j < KT; ++j) {
        float pv = 0.0f;
        if (kb + j <= qglob) pv = __expf(Ss[tid * SS_STR + j] - mx);
        lsum += pv;
        Ps[tid * PS_STR + j] = f32_to_bf16(pv);
      }
      l_s[tid] = l_s[tid] * alpha + lsum;
      m_s[tid] = mx;
      alpha_s[tid] = alpha;
    }
    __syncthreads();  // (C) P / alpha visible

    // ---- rescale O, accumulate P * V ----
    float aj[8];
#pragma unroll
    for (int j = 0; j < 8; ++j) aj[j] = alpha_s[waveMo + j + laneHalf * 8];
#pragma unroll
    for (int nt = 0; nt < 4; ++nt)
#pragma unroll
      for (int j = 0; j < 8; ++j) oacc[nt][j] *= aj[j];

    Frag16 pa;
    const unsigned short* pp = Ps + (waveMo + lane16) * PS_STR + laneHalf * 8;
    pa.u[0] = *(const u32x4*)(pp);
    pa.u[1] = *(const u32x4*)(pp + 16);
#pragma unroll
    for (int nt = 0; nt < 4; ++nt) {
      Frag16 vf;
      const unsigned short* vp = Vts + (waveNo + nt * 16 + lane16) * VS_STR + laneHalf * 8;
      vf.u[0] = *(const u32x4*)(vp);
      vf.u[1] = *(const u32x4*)(vp + 16);
      oacc[nt] = __builtin_amdgcn_wmma_f32_16x16x32_bf16(
          false, pa.v, false, vf.v, (short)0, oacc[nt], false, false);
    }
    __syncthreads();  // (D) LDS reads done before next tile overwrites
    p ^= 1;
  }

  // Final normalization and store
  float dj[8];
#pragma unroll
  for (int j = 0; j < 8; ++j) dj[j] = fast_rcp(l_s[waveMo + j + laneHalf * 8]);
#pragma unroll
  for (int nt = 0; nt < 4; ++nt) {
#pragma unroll
    for (int j = 0; j < 8; ++j) {
      int row = qbase + waveMo + j + laneHalf * 8;
      int col = waveNo + nt * 16 + lane16;
      Og[((size_t)bb * Sn + row) * rowStride + headOff + col] =
          f32_to_bf16(oacc[nt][j] * dj[j]);
    }
  }
}

// ---------------------------------------------------------------------------
// Orchestration
// ---------------------------------------------------------------------------
extern "C" void kernel_launch(void* const* d_in, const int* in_sizes, int n_in,
                              void* d_out, int out_size, void* d_ws, size_t ws_size,
                              hipStream_t stream) {
  (void)in_sizes; (void)n_in; (void)out_size; (void)ws_size;
  const int Bc = 2, Sc = 2048, Dc = 2048, Hc = 16, HDc = 128, Fc = 8192;
  const int Mc = Bc * Sc;        // 4096 rows
  const int NH = Hc * HDc;       // 2048

  const float* x    = (const float*)d_in[0];
  const float* ln1  = (const float*)d_in[1];
  const float* wq   = (const float*)d_in[2];
  const float* wk   = (const float*)d_in[3];
  const float* wv   = (const float*)d_in[4];
  const float* qln  = (const float*)d_in[5];
  const float* kln  = (const float*)d_in[6];
  const float* wo   = (const float*)d_in[7];
  const float* ln2  = (const float*)d_in[8];
  const float* wi0  = (const float*)d_in[9];
  const float* wi1  = (const float*)d_in[10];
  const float* wout = (const float*)d_in[11];
  float* out = (float*)d_out;

  char* p = (char*)d_ws;
  auto carve = [&](size_t bytes) -> char* {
    char* r = p;
    p += (bytes + 255) & ~(size_t)255;
    return r;
  };

  unsigned short* lnx_b  = (unsigned short*)carve((size_t)Mc * Dc * 2);
  unsigned short* wq_b   = (unsigned short*)carve((size_t)Dc * NH * 2);
  unsigned short* wk_b   = (unsigned short*)carve((size_t)Dc * NH * 2);
  unsigned short* wv_b   = (unsigned short*)carve((size_t)Dc * NH * 2);
  float*          q_f    = (float*)carve((size_t)Mc * NH * 4);
  float*          k_f    = (float*)carve((size_t)Mc * NH * 4);
  unsigned short* v_b    = (unsigned short*)carve((size_t)Mc * NH * 2);
  unsigned short* qn_b   = (unsigned short*)carve((size_t)Mc * NH * 2);
  unsigned short* kn_b   = (unsigned short*)carve((size_t)Mc * NH * 2);
  unsigned short* attn_b = (unsigned short*)carve((size_t)Mc * NH * 2);
  unsigned short* wo_b   = (unsigned short*)carve((size_t)NH * Dc * 2);
  float*          inter  = (float*)carve((size_t)Mc * Dc * 4);
  unsigned short* h_b    = (unsigned short*)carve((size_t)Mc * Dc * 2);
  unsigned short* wi0_b  = (unsigned short*)carve((size_t)Dc * Fc * 2);
  unsigned short* wi1_b  = (unsigned short*)carve((size_t)Dc * Fc * 2);
  float*          g0_f   = (float*)carve((size_t)Mc * Fc * 4);
  unsigned short* g_b    = (unsigned short*)carve((size_t)Mc * Fc * 2);
  unsigned short* wout_b = (unsigned short*)carve((size_t)Fc * Dc * 2);

  auto cast = [&](const float* src, unsigned short* dst, size_t n) {
    cast_f32_bf16_kernel<<<2048, 256, 0, stream>>>(src, dst, n);
  };

  // Weight casts
  cast(wq, wq_b, (size_t)Dc * NH);
  cast(wk, wk_b, (size_t)Dc * NH);
  cast(wv, wv_b, (size_t)Dc * NH);
  cast(wo, wo_b, (size_t)NH * Dc);
  cast(wi0, wi0_b, (size_t)Dc * Fc);
  cast(wi1, wi1_b, (size_t)Dc * Fc);
  cast(wout, wout_b, (size_t)Fc * Dc);

  // ---- pre-attention RMSNorm ----
  rmsnorm_row_kernel<<<Mc, 256, 0, stream>>>(x, ln1, lnx_b, Dc);

  // ---- QKV projections ----
  dim3 gQKV(NH / TBN, Mc / TBM);
  gemm_bf16_wmma<<<gQKV, 256, 0, stream>>>(lnx_b, wq_b, nullptr, q_f, Mc, NH, Dc, 0);
  gemm_bf16_wmma<<<gQKV, 256, 0, stream>>>(lnx_b, wk_b, nullptr, k_f, Mc, NH, Dc, 0);
  gemm_bf16_wmma<<<gQKV, 256, 0, stream>>>(lnx_b, wv_b, nullptr, v_b, Mc, NH, Dc, 1);

  // ---- per-head QK RMSNorm ----
  headnorm_kernel<<<Mc * Hc, 128, 0, stream>>>(q_f, qln, qn_b);
  headnorm_kernel<<<Mc * Hc, 128, 0, stream>>>(k_f, kln, kn_b);

  // ---- causal flash attention ----
  dim3 gAttn(Bc * Hc, Sc / QT);
  flash_attn_wmma<<<gAttn, 256, 0, stream>>>(qn_b, kn_b, v_b, attn_b, Bc, Sc, Hc, HDc);

  // ---- output projection + residual: inter = x + attn @ wo ----
  dim3 gWo(Dc / TBN, Mc / TBM);
  gemm_bf16_wmma<<<gWo, 256, 0, stream>>>(attn_b, wo_b, x, inter, Mc, Dc, NH, 3);

  // ---- pre-MLP RMSNorm ----
  rmsnorm_row_kernel<<<Mc, 256, 0, stream>>>(inter, ln2, h_b, Dc);

  // ---- gated MLP ----
  dim3 gMlp(Fc / TBN, Mc / TBM);
  gemm_bf16_wmma<<<gMlp, 256, 0, stream>>>(h_b, wi0_b, nullptr, g0_f, Mc, Fc, Dc, 0);
  gemm_bf16_wmma<<<gMlp, 256, 0, stream>>>(h_b, wi1_b, g0_f, g_b, Mc, Fc, Dc, 2);

  // ---- down projection + residual: out = inter + g @ wout ----
  dim3 gOut(Dc / TBN, Mc / TBM);
  gemm_bf16_wmma<<<gOut, 256, 0, stream>>>(g_b, wout_b, inter, out, Mc, Dc, Fc, 3);
}
